// MultiheadAttention_14688788152784
// MI455X (gfx1250) — compile-verified, matrix operand order checked by reading, not run on hardware
//
#include <hip/hip_runtime.h>
#include <hip/hip_bf16.h>

// Problem constants (match reference)
#define SDIM 1024
#define BDIM 8
#define EDIM 1024
#define HNUM 16
#define HD   64
#define NTOK (SDIM * BDIM)   // 8192 tokens
#define E3   (3 * EDIM)      // 3072

typedef __attribute__((ext_vector_type(16))) _Float16 v16h;
typedef __attribute__((ext_vector_type(8)))  _Float16 v8h;
typedef __attribute__((ext_vector_type(8)))  float    v8f;

// ---------------------------------------------------------------------------
// Fragment helpers (CDNA5 wave32 WMMA 16x16x32 f16 layouts, ISA 7.12.2)
// A (16x32): lane L -> row m = L&15; halves at K=(L>>4)*8 + {0..7, 16..23}.
// B (32x16) is the mirror (lane -> column), same K map -> identical
// addressing when the source matrix is row-major with K contiguous.
// ---------------------------------------------------------------------------
static __device__ inline v16h load_frag16(const _Float16* p, int ld, int lane) {
    const int m  = lane & 15;
    const int kb = (lane >> 4) << 3;
    const _Float16* r = p + (size_t)m * (size_t)ld + kb;
    v8h lo = *(const v8h*)(r);
    v8h hi = *(const v8h*)(r + 16);
    return __builtin_shufflevector(lo, hi, 0, 1, 2, 3, 4, 5, 6, 7,
                                           8, 9, 10, 11, 12, 13, 14, 15);
}

static __device__ inline v8f wmma16(v16h a, v16h b, v8f c) {
    // 8-arg form: (neg_a, A, neg_b, B, c_mod, C, reuse_a, reuse_b)
    return __builtin_amdgcn_wmma_f32_16x16x32_f16(false, a, false, b,
                                                  (short)0, c, false, false);
}

static __device__ inline v8f vzero8() {
    v8f z = {0.f, 0.f, 0.f, 0.f, 0.f, 0.f, 0.f, 0.f};
    return z;
}

// Async global->LDS copy of one 16-byte chunk per lane (ASYNCcnt-tracked).
static __device__ inline void async_copy_b128(unsigned lds_byte_off,
                                              const _Float16* gaddr) {
    unsigned long long ga = (unsigned long long)(uintptr_t)gaddr;
    asm volatile("global_load_async_to_lds_b128 %0, %1, off"
                 :: "v"(lds_byte_off), "v"(ga) : "memory");
}

static __device__ inline void wait_async0() {
    asm volatile("s_wait_asynccnt 0x0" ::: "memory");
}

// ---------------------------------------------------------------------------
// f32 -> f16 conversion
// ---------------------------------------------------------------------------
__global__ void f32_to_f16_kernel(const float* __restrict__ src,
                                  _Float16* __restrict__ dst, int n) {
    int i = blockIdx.x * blockDim.x + threadIdx.x;
    if (i < n) dst[i] = (_Float16)src[i];
}

// ---------------------------------------------------------------------------
// Shared-memory W-tile staging used by both projection GEMMs.
// Tile = 64 output columns x 32 k, re+im (8KB), row stride 40 halves so the
// 16-byte fragment chunks stay aligned and rows map to distinct LDS banks.
// 512 x 16B chunks staged by 128 threads -> 4 async b128 per thread.
// ---------------------------------------------------------------------------
#define WROW 40

static __device__ inline void stage_w_tile(_Float16* wbuf,   // [2][64][WROW]
                                           const _Float16* wr,
                                           const _Float16* wi,
                                           int col0, int kb, int tid) {
#pragma unroll
    for (int i = 0; i < 4; ++i) {
        const int q   = tid + i * 128;        // 0..511
        const int mat = q >> 8;               // 0 = re, 1 = im
        const int rr  = q & 255;
        const int c   = rr >> 2;              // column within tile, 0..63
        const int ck  = (rr & 3) << 3;        // k-chunk start (halves)
        const _Float16* g = (mat ? wi : wr) + (size_t)(col0 + c) * EDIM + kb + ck;
        unsigned lds = (unsigned)(uintptr_t)(wbuf + (mat * 64 + c) * WROW + ck);
        async_copy_b128(lds, g);
    }
}

// ---------------------------------------------------------------------------
// QKV complex projection:  y = x @ W^T + b, writing q/k/v head-major
// (B,H,S,HD) f16.  Block = 128 threads (4 waves); wave = 16x64 complex tile;
// W tile staged once per block via async-to-LDS.
// ---------------------------------------------------------------------------
__global__ void __launch_bounds__(128) qkv_gemm_kernel(
    const _Float16* __restrict__ xqr, const _Float16* __restrict__ xqi,
    const _Float16* __restrict__ xkr, const _Float16* __restrict__ xki,
    const _Float16* __restrict__ xvr, const _Float16* __restrict__ xvi,
    const _Float16* __restrict__ wr,  const _Float16* __restrict__ wi,
    const float* __restrict__ bre,    const float* __restrict__ bim,
    _Float16* __restrict__ Qr, _Float16* __restrict__ Qi,
    _Float16* __restrict__ Kr, _Float16* __restrict__ Ki,
    _Float16* __restrict__ Vr, _Float16* __restrict__ Vi)
{
    __shared__ __align__(16) _Float16 wbuf[2 * 64 * WROW];

    const int lane = threadIdx.x & 31;
    const int wave = threadIdx.x >> 5;
    const int row0 = (blockIdx.x * 4 + wave) * 16;   // token tile
    const int col0 = blockIdx.y * 64;                // output-feature tile
    const int sel  = col0 >> 10;                     // 0=q 1=k 2=v

    const _Float16* Xr = (sel == 0) ? xqr : ((sel == 1) ? xkr : xvr);
    const _Float16* Xi = (sel == 0) ? xqi : ((sel == 1) ? xki : xvi);

    v8f accR[4], accI[4];
#pragma unroll
    for (int j = 0; j < 4; ++j) { accR[j] = vzero8(); accI[j] = vzero8(); }

    for (int kb = 0; kb < EDIM; kb += 32) {
        __syncthreads();   // all waves done reading previous W tile
        stage_w_tile(wbuf, wr, wi, col0, kb, (int)threadIdx.x);

        v16h ar  = load_frag16(Xr + (size_t)row0 * EDIM + kb, EDIM, lane);
        v16h ai  = load_frag16(Xi + (size_t)row0 * EDIM + kb, EDIM, lane);
        v16h nai = -ai;  // f16 WMMA has no A/B NEG modifier; negate explicitly
        if (kb + 32 < EDIM) {   // prefetch next A tiles (global_prefetch_b8)
            __builtin_prefetch(Xr + (size_t)row0 * EDIM + kb + 32, 0, 3);
            __builtin_prefetch(Xi + (size_t)row0 * EDIM + kb + 32, 0, 3);
        }

        wait_async0();
        __syncthreads();   // W tile visible to all waves

#pragma unroll
        for (int j = 0; j < 4; ++j) {
            v16h brf = load_frag16(wbuf + (0 * 64 + j * 16) * WROW, WROW, lane);
            v16h bif = load_frag16(wbuf + (1 * 64 + j * 16) * WROW, WROW, lane);
            accR[j] = wmma16(ar,  brf, accR[j]);
            accR[j] = wmma16(nai, bif, accR[j]);   // - ai*bi
            accI[j] = wmma16(ar,  bif, accI[j]);
            accI[j] = wmma16(ai,  brf, accI[j]);
        }
    }

    _Float16* Dr = (sel == 0) ? Qr : ((sel == 1) ? Kr : Vr);
    _Float16* Di = (sel == 0) ? Qi : ((sel == 1) ? Ki : Vi);

#pragma unroll
    for (int j = 0; j < 4; ++j) {
        const int col = col0 + j * 16 + (lane & 15);
        const float br_v = bre[col];
        const float bi_v = bim[col];
        const int e  = col & (EDIM - 1);
        const int hh = e >> 6;         // head
        const int d  = e & 63;         // dim in head
#pragma unroll
        for (int r = 0; r < 8; ++r) {
            const int tok = row0 + ((lane >> 4) << 3) + r;   // C layout rows
            const int s = tok >> 3;        // token = s*B + b, B=8
            const int b = tok & 7;
            const size_t off = (((size_t)(b * HNUM + hh) * SDIM + s) * HD) + d;
            Dr[off] = (_Float16)(accR[j][r] + br_v);
            Di[off] = (_Float16)(accI[j][r] + bi_v);
        }
    }
}

// ---------------------------------------------------------------------------
// Flash attention per (b,h):  scores = (qr.kr^T + qi.ki^T)/8, online softmax,
// out = P @ v (re and im).  Block = 4 waves, each wave owns 16 query rows.
// V tiles staged transposed in LDS; P bounces through LDS (C- -> A-layout).
// ---------------------------------------------------------------------------
__global__ void __launch_bounds__(128) attn_kernel(
    const _Float16* __restrict__ Qr, const _Float16* __restrict__ Qi,
    const _Float16* __restrict__ Kr, const _Float16* __restrict__ Ki,
    const _Float16* __restrict__ Vr, const _Float16* __restrict__ Vi,
    _Float16* __restrict__ Or, _Float16* __restrict__ Oi)
{
    __shared__ __align__(16) _Float16 vtr[64][40];       // V^T re, padded rows
    __shared__ __align__(16) _Float16 vti[64][40];       // V^T im
    __shared__ __align__(16) _Float16 pbuf[4][16][32];   // per-wave P tile

    const int lane = threadIdx.x & 31;
    const int wave = threadIdx.x >> 5;
    const int bh = blockIdx.y;          // = b*H + h
    const int b  = bh >> 4;
    const int h  = bh & 15;
    const int s0 = blockIdx.x * 64 + wave * 16;

    const size_t base = (size_t)bh * SDIM * HD;

    v16h aqr[2], aqi[2];
#pragma unroll
    for (int ks = 0; ks < 2; ++ks) {
        aqr[ks] = load_frag16(Qr + base + (size_t)s0 * HD + ks * 32, HD, lane);
        aqi[ks] = load_frag16(Qi + base + (size_t)s0 * HD + ks * 32, HD, lane);
    }

    v8f oR[4], oI[4];
#pragma unroll
    for (int j = 0; j < 4; ++j) { oR[j] = vzero8(); oI[j] = vzero8(); }
    float mrun[8], lrun[8];
#pragma unroll
    for (int r = 0; r < 8; ++r) { mrun[r] = -1e30f; lrun[r] = 0.f; }

    const float scale = 0.125f;   // 1/sqrt(64)

    for (int t0 = 0; t0 < SDIM; t0 += 32) {
        __syncthreads();   // protect previous iteration's V-tile reads
        // Cooperative staging of V[t0..t0+31][0..63] transposed into LDS
#pragma unroll 4
        for (int i = 0; i < 16; ++i) {
            const int idx = (int)threadIdx.x + i * 128;   // 0..2047
            const int t = idx >> 6;
            const int d = idx & 63;
            vtr[d][t] = Vr[base + (size_t)(t0 + t) * HD + d];
            vti[d][t] = Vi[base + (size_t)(t0 + t) * HD + d];
        }
        __syncthreads();

        // Scores: 16x32 block as two 16x16 C tiles
        v8f sc[2];
        sc[0] = vzero8(); sc[1] = vzero8();
#pragma unroll
        for (int ts = 0; ts < 2; ++ts) {
#pragma unroll
            for (int ks = 0; ks < 2; ++ks) {
                v16h bkr = load_frag16(Kr + base + (size_t)(t0 + ts * 16) * HD + ks * 32, HD, lane);
                v16h bki = load_frag16(Ki + base + (size_t)(t0 + ts * 16) * HD + ks * 32, HD, lane);
                sc[ts] = wmma16(aqr[ks], bkr, sc[ts]);
                sc[ts] = wmma16(aqi[ks], bki, sc[ts]);
            }
        }

        // Online softmax across the 32 new columns (butterfly within 16-lane
        // half-groups: C-layout spreads each row over 16 lanes)
#pragma unroll
        for (int r = 0; r < 8; ++r) {
            float x0 = sc[0][r] * scale;
            float x1 = sc[1][r] * scale;
            float mx = fmaxf(x0, x1);
            mx = fmaxf(mx, __shfl_xor(mx, 1, 32));
            mx = fmaxf(mx, __shfl_xor(mx, 2, 32));
            mx = fmaxf(mx, __shfl_xor(mx, 4, 32));
            mx = fmaxf(mx, __shfl_xor(mx, 8, 32));
            const float mnew = fmaxf(mrun[r], mx);
            const float corr = __expf(mrun[r] - mnew);
            const float p0 = __expf(x0 - mnew);
            const float p1 = __expf(x1 - mnew);
            float rs = p0 + p1;
            rs += __shfl_xor(rs, 1, 32);
            rs += __shfl_xor(rs, 2, 32);
            rs += __shfl_xor(rs, 4, 32);
            rs += __shfl_xor(rs, 8, 32);
            lrun[r] = lrun[r] * corr + rs;
            mrun[r] = mnew;
#pragma unroll
            for (int j = 0; j < 4; ++j) { oR[j][r] *= corr; oI[j][r] *= corr; }
            const int row = ((lane >> 4) << 3) + r;
            pbuf[wave][row][(lane & 15)]      = (_Float16)p0;
            pbuf[wave][row][16 + (lane & 15)] = (_Float16)p1;
        }

        // LDS pipeline is in-order per wave; fence explicitly anyway
        asm volatile("s_wait_dscnt 0x0" ::: "memory");

        // Re-read P in A-fragment layout, then P @ V  (re and im)
        v16h pa = load_frag16(&pbuf[wave][0][0], 32, lane);
#pragma unroll
        for (int j = 0; j < 4; ++j) {
            v16h bvr = load_frag16(&vtr[j * 16][0], 40, lane);
            v16h bvi = load_frag16(&vti[j * 16][0], 40, lane);
            oR[j] = wmma16(pa, bvr, oR[j]);
            oI[j] = wmma16(pa, bvi, oI[j]);
        }
    }

    // Normalize and write (S,B,E) f16 for the output projection
#pragma unroll
    for (int j = 0; j < 4; ++j) {
        const int d = j * 16 + (lane & 15);
#pragma unroll
        for (int r = 0; r < 8; ++r) {
            const int row = ((lane >> 4) << 3) + r;
            const int s = s0 + row;
            const float invl = 1.0f / lrun[r];
            const size_t off = ((size_t)(s * BDIM + b)) * EDIM + h * HD + d;
            Or[off] = (_Float16)(oR[j][r] * invl);
            Oi[off] = (_Float16)(oI[j][r] * invl);
        }
    }
}

// ---------------------------------------------------------------------------
// Output complex projection:  y = out @ Wo^T + bo -> f32 d_out stacked
// (2, S, B, E).  Same async-staged structure as the QKV GEMM.
// ---------------------------------------------------------------------------
__global__ void __launch_bounds__(128) out_gemm_kernel(
    const _Float16* __restrict__ Ar, const _Float16* __restrict__ Ai,
    const _Float16* __restrict__ wr, const _Float16* __restrict__ wi,
    const float* __restrict__ bre,   const float* __restrict__ bim,
    float* __restrict__ out)
{
    __shared__ __align__(16) _Float16 wbuf[2 * 64 * WROW];

    const int lane = threadIdx.x & 31;
    const int wave = threadIdx.x >> 5;
    const int row0 = (blockIdx.x * 4 + wave) * 16;
    const int col0 = blockIdx.y * 64;

    v8f accR[4], accI[4];
#pragma unroll
    for (int j = 0; j < 4; ++j) { accR[j] = vzero8(); accI[j] = vzero8(); }

    for (int kb = 0; kb < EDIM; kb += 32) {
        __syncthreads();
        stage_w_tile(wbuf, wr, wi, col0, kb, (int)threadIdx.x);

        v16h ar  = load_frag16(Ar + (size_t)row0 * EDIM + kb, EDIM, lane);
        v16h ai  = load_frag16(Ai + (size_t)row0 * EDIM + kb, EDIM, lane);
        v16h nai = -ai;
        if (kb + 32 < EDIM) {
            __builtin_prefetch(Ar + (size_t)row0 * EDIM + kb + 32, 0, 3);
            __builtin_prefetch(Ai + (size_t)row0 * EDIM + kb + 32, 0, 3);
        }

        wait_async0();
        __syncthreads();

#pragma unroll
        for (int j = 0; j < 4; ++j) {
            v16h brf = load_frag16(wbuf + (0 * 64 + j * 16) * WROW, WROW, lane);
            v16h bif = load_frag16(wbuf + (1 * 64 + j * 16) * WROW, WROW, lane);
            accR[j] = wmma16(ar,  brf, accR[j]);
            accR[j] = wmma16(nai, bif, accR[j]);
            accI[j] = wmma16(ar,  bif, accI[j]);
            accI[j] = wmma16(ai,  brf, accI[j]);
        }
    }

    const size_t imOff = (size_t)NTOK * EDIM;
#pragma unroll
    for (int j = 0; j < 4; ++j) {
        const int col = col0 + j * 16 + (lane & 15);
        const float br_v = bre[col];
        const float bi_v = bim[col];
#pragma unroll
        for (int r = 0; r < 8; ++r) {
            const int tok = row0 + ((lane >> 4) << 3) + r;
            const size_t off = (size_t)tok * EDIM + col;
            out[off]         = accR[j][r] + br_v;
            out[imOff + off] = accI[j][r] + bi_v;
        }
    }
}

// ---------------------------------------------------------------------------
// Host launcher
// ---------------------------------------------------------------------------
extern "C" void kernel_launch(void* const* d_in, const int* in_sizes, int n_in,
                              void* d_out, int out_size, void* d_ws, size_t ws_size,
                              hipStream_t stream) {
    (void)in_sizes; (void)n_in; (void)out_size; (void)ws_size;

    const size_t NE   = (size_t)NTOK * EDIM;   // 8,388,608
    const size_t WQKV = (size_t)E3 * EDIM;     // 3,145,728
    const size_t WO   = (size_t)EDIM * EDIM;   // 1,048,576

    _Float16* hws = (_Float16*)d_ws;
    _Float16* xqr = hws;            _Float16* xqi = xqr + NE;
    _Float16* xkr = xqi + NE;       _Float16* xki = xkr + NE;
    _Float16* xvr = xki + NE;       _Float16* xvi = xvr + NE;
    _Float16* wr  = xvi + NE;       _Float16* wi  = wr + WQKV;
    _Float16* wor = wi + WQKV;      _Float16* woi = wor + WO;
    _Float16* Qr  = woi + WO;       _Float16* Qi  = Qr + NE;
    _Float16* Kr  = Qi + NE;        _Float16* Ki  = Kr + NE;
    _Float16* Vr  = Ki + NE;        _Float16* Vi  = Vr + NE;
    _Float16* Aor = Vi + NE;        _Float16* Aoi = Aor + NE;

    auto cvt = [&](const void* src, _Float16* dst, size_t n) {
        f32_to_f16_kernel<<<dim3((unsigned)((n + 255) / 256)), dim3(256), 0, stream>>>(
            (const float*)src, dst, (int)n);
    };

    // Activations and weights -> f16
    cvt(d_in[0], xqr, NE);  cvt(d_in[1], xqi, NE);
    cvt(d_in[2], xkr, NE);  cvt(d_in[3], xki, NE);
    cvt(d_in[4], xvr, NE);  cvt(d_in[5], xvi, NE);
    cvt(d_in[6], wr, WQKV); cvt(d_in[7], wi, WQKV);
    cvt(d_in[10], wor, WO); cvt(d_in[11], woi, WO);

    // QKV projection (complex), head-major f16 outputs
    {
        dim3 grid(NTOK / 64, E3 / 64);
        qkv_gemm_kernel<<<grid, dim3(128), 0, stream>>>(
            xqr, xqi, xkr, xki, xvr, xvi, wr, wi,
            (const float*)d_in[8], (const float*)d_in[9],
            Qr, Qi, Kr, Ki, Vr, Vi);
    }

    // Flash attention per (b,h)
    {
        dim3 grid(SDIM / 64, BDIM * HNUM);
        attn_kernel<<<grid, dim3(128), 0, stream>>>(Qr, Qi, Kr, Ki, Vr, Vi, Aor, Aoi);
    }

    // Output projection (complex), f32 result stacked (re, im)
    {
        dim3 grid(NTOK / 64, EDIM / 64);
        out_gemm_kernel<<<grid, dim3(128), 0, stream>>>(
            Aor, Aoi, wor, woi,
            (const float*)d_in[12], (const float*)d_in[13],
            (float*)d_out);
    }
}
